// MultiHeadAttention_47940424957999
// MI455X (gfx1250) — matrix operand order repaired and checked by reading, not compiled, not run on hardware
//
#include <hip/hip_runtime.h>
#include <hip/hip_bf16.h>

// MHA: B=8, N=2048, D=256, H=8, DH=32.  All WMMA f16 w/ f32 accumulate.
typedef __attribute__((ext_vector_type(16))) _Float16 v16h;
typedef __attribute__((ext_vector_type(8)))  float    v8f;
typedef __attribute__((ext_vector_type(4)))  unsigned int v4u;
typedef __attribute__((ext_vector_type(8)))  unsigned int v8u;

// ---- CDNA5 WMMA fragment index helpers (ISA 7.12.2, wave32) ----
// A-matrix (16x32 f16): lane L holds row M=L&15; half h -> K:
//   lanes 0-15 : K = {0..7, 16..23};  lanes 16-31 : K = {8..15, 24..31}
static __device__ __forceinline__ int a_k(int L, int h) {
  return ((h & 8) << 1) + ((L & 16) >> 1) + (h & 7);
}
// B-matrix (32x16 f16): lane L holds col N=L&15; half h -> K:
//   lanes 0-15 : K = {0..7, 16..23};  lanes 16-31 : K = {8..15, 24..31}
static __device__ __forceinline__ int b_k(int L, int h) {
  return ((h & 8) << 1) + ((L & 16) >> 1) + (h & 7);
}
// C/D (16x16 f32): VGPR r, lane L -> row = r + ((L&16)>>1), col = L&15

// =====================================================================
// Kernel 1: qkv = concat(q,k,v) @ Wqkv^T + bqkv, scattered to head-major
// f16 tensors Qh/Kh/Vh [B,H,N,32].  M=16384, Nout=768, K=768.
// Block: 256 thr (8 waves), tile 128x64, each wave 32x32 (2x2 WMMA).
// =====================================================================
__global__ __launch_bounds__(256)
void qkv_gemm_kernel(const float* __restrict__ q, const float* __restrict__ k,
                     const float* __restrict__ v, const float* __restrict__ W,
                     const float* __restrict__ bias,
                     _Float16* __restrict__ Qh, _Float16* __restrict__ Kh,
                     _Float16* __restrict__ Vh)
{
  __shared__ __align__(16) _Float16 As[128][40];   // A tile 128x32 (padded)
  __shared__ __align__(16) _Float16 Bs[64][32];    // B tile as [n][kk]
  const int t  = threadIdx.x;
  const int L  = t & 31;
  const int wv = t >> 5;
  const int wr = wv & 3;          // wave row group (0..3)
  const int wc = wv >> 2;         // wave col group (0..1)
  const int m0 = blockIdx.y * 128;
  const int n0 = blockIdx.x * 64;

  v8f acc[2][2] = {};

  for (int kt = 0; kt < 24; ++kt) {
    const int f0 = kt * 32;
    const float* xsrc = (f0 < 256) ? q : (f0 < 512) ? k : v;
    const int fo = f0 & 255;
    // stage A tile (fp32 -> f16)
    #pragma unroll
    for (int e = 0; e < 16; ++e) {
      int idx = t + e * 256;
      int r = idx >> 5, c = idx & 31;
      As[r][c] = (_Float16)xsrc[(size_t)(m0 + r) * 256 + fo + c];
    }
    // stage B tile: Bs[n][kk] = Wqkv[(n0+n)*768 + f0+kk]
    #pragma unroll
    for (int e = 0; e < 8; ++e) {
      int idx = t + e * 256;
      int bn = idx >> 5, kk = idx & 31;
      Bs[bn][kk] = (_Float16)W[(size_t)(n0 + bn) * 768 + f0 + kk];
    }
    __syncthreads();

    v16h af[2], bf[2];
    #pragma unroll
    for (int i = 0; i < 2; ++i) {
      int row = wr * 32 + i * 16 + (L & 15);
      #pragma unroll
      for (int h = 0; h < 16; ++h) af[i][h] = As[row][a_k(L, h)];
    }
    #pragma unroll
    for (int j = 0; j < 2; ++j) {
      int bn = wc * 32 + j * 16 + (L & 15);
      #pragma unroll
      for (int h = 0; h < 16; ++h) bf[j][h] = Bs[bn][b_k(L, h)];
    }
    #pragma unroll
    for (int i = 0; i < 2; ++i)
      #pragma unroll
      for (int j = 0; j < 2; ++j)
        acc[i][j] = __builtin_amdgcn_wmma_f32_16x16x32_f16(
            false, af[i], false, bf[j], (short)0, acc[i][j], false, false);
    __syncthreads();
  }

  // epilogue: add bias, scatter to head-major f16
  #pragma unroll
  for (int i = 0; i < 2; ++i) {
    #pragma unroll
    for (int j = 0; j < 2; ++j) {
      #pragma unroll
      for (int r = 0; r < 8; ++r) {
        int row = wr * 32 + i * 16 + r + ((L & 16) >> 1);
        int m = m0 + row;
        int o = n0 + wc * 32 + j * 16 + (L & 15);
        float val = acc[i][j][r] + bias[o];
        int bb = m >> 11, nn = m & 2047;
        int head = (o & 255) >> 5, dh = o & 31;
        size_t dst = (((size_t)(bb * 8 + head) * 2048) + nn) * 32 + dh;
        _Float16 hv = (_Float16)val;
        if (o < 256)      Qh[dst] = hv;
        else if (o < 512) Kh[dst] = hv;
        else              Vh[dst] = hv;
      }
    }
  }
}

// =====================================================================
// Kernel 2: causal flash attention per (b,h).  Block = 4 waves = 64 rows;
// each wave owns a 16-row Q tile, loops over 32-key blocks staged in LDS.
// K tile staged by the Tensor Data Mover (tensor_load_to_lds, TENSORcnt);
// V tile staged by async LDS DMA (global_load_async_to_lds_b128, ASYNCcnt).
// scores = (Q K^T) * sqrt(32), causal fill -32767, online softmax.
// =====================================================================
__global__ __launch_bounds__(128)
void flash_attn_kernel(const _Float16* __restrict__ Qh,
                       const _Float16* __restrict__ Kh,
                       const _Float16* __restrict__ Vh,
                       _Float16* __restrict__ Oh)
{
  __shared__ __align__(16) _Float16 Kt[1024];      // 32 keys x 32 dh
  __shared__ __align__(16) _Float16 Vt[1024];      // 32 keys x 32 dh
  __shared__ __align__(16) _Float16 Pt[4][512];    // per-wave P 16x32
  const int t  = threadIdx.x;
  const int L  = t & 31;
  const int wv = t >> 5;
  const int bh = blockIdx.x;
  const int row0 = blockIdx.y * 64 + wv * 16;
  const size_t base = (size_t)bh * 2048 * 32;

  const unsigned ktLds = (unsigned)(unsigned long long)(const void*)&Kt[0];
  const unsigned vtLds = (unsigned)(unsigned long long)(const void*)&Vt[0];

  // Q A-fragment (16x32), scattered u16 loads per documented layout
  v16h qf;
  {
    const _Float16* qp = Qh + base + (size_t)(row0 + (L & 15)) * 32;
    #pragma unroll
    for (int h = 0; h < 16; ++h) qf[h] = qp[a_k(L, h)];
  }

  v8f o0 = {}, o1 = {};
  float mrow[8], lrow[8];
  #pragma unroll
  for (int r = 0; r < 8; ++r) { mrow[r] = -1e30f; lrow[r] = 0.f; }

  const int colg = L & 15;
  const int jmax = blockIdx.y * 64 + 63;           // causal upper bound (block)
  for (int j0 = 0; j0 <= jmax; j0 += 32) {
    // ---- stage K tile (2KB, contiguous) via TDM, wave 0 only ----
    if (wv == 0) {
      unsigned long long kga =
          (unsigned long long)(Kh + base + (size_t)j0 * 32);
      // D# group 0: count=1 | lds_addr | global_addr[56:0] | type=2
      v4u g0;
      g0[0] = 1u;                                    // count=1, user mode
      g0[1] = ktLds;                                 // lds_addr (bytes)
      g0[2] = (unsigned)(kga & 0xffffffffu);         // global_addr[31:0]
      g0[3] = (unsigned)((kga >> 32) & 0x01ffffffu)  // global_addr[56:32]
              | (2u << 30);                          // type = 2 (image)
      // D# group 1: data_size=2B, 1-row tile of 1024 elements
      v8u g1;
      g1[0] = (1u << 16);                            // data_size=1 (2 bytes)
      g1[1] = (1024u & 0xffffu) << 16;               // tensor_dim0 = 1024
      g1[2] = (1u << 16);                            // tensor_dim1 = 1
      g1[3] = (1024u << 16);                         // tile_dim0 = 1024
      g1[4] = 1u;                                    // tile_dim1 = 1
      g1[5] = 1024u;                                 // tensor_dim0_stride
      g1[6] = (1024u & 0xffffu) << 16;               // tensor_dim1_stride
      g1[7] = 0u;
      asm volatile("tensor_load_to_lds %0, %1"
                   :: "s"(g0), "s"(g1) : "memory");
      __builtin_amdgcn_s_wait_tensorcnt(0);
    }
    // ---- stage V tile via async global->LDS DMA (16B per lane) ----
    {
      unsigned long long vga =
          (unsigned long long)(Vh + base + (size_t)j0 * 32) +
          (unsigned long long)t * 16ull;
      unsigned vdst = vtLds + (unsigned)t * 16u;
      asm volatile("global_load_async_to_lds_b128 %0, %1, off"
                   :: "v"(vdst), "v"(vga) : "memory");
      asm volatile("s_wait_asynccnt 0x0" ::: "memory");
    }
    __syncthreads();

    // K^T B-fragments: col = key, K-dim = dh (lanes 0-15: dh {0..7,16..23}; 16-31: {8..15,24..31})
    v16h kf0, kf1;
    #pragma unroll
    for (int h = 0; h < 16; ++h) {
      int kk = b_k(L, h);                          // dh 0..31
      kf0[h] = Kt[(colg) * 32 + kk];
      kf1[h] = Kt[(16 + colg) * 32 + kk];
    }
    v8f zero = {};
    v8f s0 = __builtin_amdgcn_wmma_f32_16x16x32_f16(false, qf, false, kf0,
                                                    (short)0, zero, false, false);
    v8f s1 = __builtin_amdgcn_wmma_f32_16x16x32_f16(false, qf, false, kf1,
                                                    (short)0, zero, false, false);

    // scale (sqrt(DH)=sqrt(32)) + causal mask with reference fill value
    const float scale = 5.656854249f;
    #pragma unroll
    for (int r = 0; r < 8; ++r) {
      int row = row0 + r + ((L & 16) >> 1);
      float a = s0[r] * scale;
      float b = s1[r] * scale;
      if (j0 + colg > row)      a = -32767.0f;
      if (j0 + 16 + colg > row) b = -32767.0f;
      s0[r] = a; s1[r] = b;
    }

    // online softmax: row reductions across the 16-lane half-groups
    float alpha[8];
    #pragma unroll
    for (int r = 0; r < 8; ++r) {
      float mx = fmaxf(s0[r], s1[r]);
      #pragma unroll
      for (int off = 8; off; off >>= 1) mx = fmaxf(mx, __shfl_xor(mx, off, 16));
      float mnew = fmaxf(mrow[r], mx);
      alpha[r] = __expf(mrow[r] - mnew);
      float p0 = __expf(s0[r] - mnew);
      float p1 = __expf(s1[r] - mnew);
      float rs = p0 + p1;
      #pragma unroll
      for (int off = 8; off; off >>= 1) rs += __shfl_xor(rs, off, 16);
      lrow[r] = lrow[r] * alpha[r] + rs;
      mrow[r] = mnew;
      s0[r] = p0; s1[r] = p1;
      o0[r] *= alpha[r]; o1[r] *= alpha[r];
    }

    // re-layout P (C/D layout -> A layout) via per-wave LDS
    _Float16* P = Pt[wv];
    #pragma unroll
    for (int r = 0; r < 8; ++r) {
      int row = r + ((L & 16) >> 1);
      P[row * 32 + colg]      = (_Float16)s0[r];
      P[row * 32 + 16 + colg] = (_Float16)s1[r];
    }
    asm volatile("s_wait_dscnt 0" ::: "memory");   // intra-wave LDS RAW fence

    v16h pf;
    {
      int row = L & 15;
      #pragma unroll
      for (int h = 0; h < 16; ++h) pf[h] = P[row * 32 + a_k(L, h)];
    }
    // V B-fragments: col = dh, K-dim = key
    v16h vf0, vf1;
    #pragma unroll
    for (int h = 0; h < 16; ++h) {
      int kk = b_k(L, h);                          // key 0..31
      vf0[h] = Vt[kk * 32 + colg];
      vf1[h] = Vt[kk * 32 + 16 + colg];
    }
    o0 = __builtin_amdgcn_wmma_f32_16x16x32_f16(false, pf, false, vf0,
                                                (short)0, o0, false, false);
    o1 = __builtin_amdgcn_wmma_f32_16x16x32_f16(false, pf, false, vf1,
                                                (short)0, o1, false, false);
    __syncthreads();
  }

  // normalize + store head-major f16 accumulator
  #pragma unroll
  for (int r = 0; r < 8; ++r) {
    float inv = 1.0f / lrow[r];
    int row = row0 + r + ((L & 16) >> 1);
    size_t dst = base + (size_t)row * 32;
    Oh[dst + colg]      = (_Float16)(o0[r] * inv);
    Oh[dst + 16 + colg] = (_Float16)(o1[r] * inv);
  }
}

// =====================================================================
// Kernel 3: out = gather(Oh) @ Wout^T + bout.  M=16384, Nout=256, K=256.
// =====================================================================
__global__ __launch_bounds__(256)
void out_proj_kernel(const _Float16* __restrict__ Oh, const float* __restrict__ W,
                     const float* __restrict__ bias, float* __restrict__ out)
{
  __shared__ __align__(16) _Float16 As[128][40];
  __shared__ __align__(16) _Float16 Bs[64][32];
  const int t  = threadIdx.x;
  const int L  = t & 31;
  const int wv = t >> 5;
  const int wr = wv & 3;
  const int wc = wv >> 2;
  const int m0 = blockIdx.y * 128;
  const int n0 = blockIdx.x * 64;

  v8f acc[2][2] = {};

  for (int kt = 0; kt < 8; ++kt) {
    const int k0 = kt * 32;
    // A tile: gather head-interleaved f16 accumulator
    #pragma unroll
    for (int e = 0; e < 16; ++e) {
      int idx = t + e * 256;
      int r = idx >> 5, c = idx & 31;
      int m = m0 + r, d = k0 + c;
      int bb = m >> 11, nn = m & 2047;
      As[r][c] = Oh[(((size_t)(bb * 8 + (d >> 5)) * 2048) + nn) * 32 + (d & 31)];
    }
    #pragma unroll
    for (int e = 0; e < 8; ++e) {
      int idx = t + e * 256;
      int bn = idx >> 5, kk = idx & 31;
      Bs[bn][kk] = (_Float16)W[(size_t)(n0 + bn) * 256 + k0 + kk];
    }
    __syncthreads();

    v16h af[2], bf[2];
    #pragma unroll
    for (int i = 0; i < 2; ++i) {
      int row = wr * 32 + i * 16 + (L & 15);
      #pragma unroll
      for (int h = 0; h < 16; ++h) af[i][h] = As[row][a_k(L, h)];
    }
    #pragma unroll
    for (int j = 0; j < 2; ++j) {
      int bn = wc * 32 + j * 16 + (L & 15);
      #pragma unroll
      for (int h = 0; h < 16; ++h) bf[j][h] = Bs[bn][b_k(L, h)];
    }
    #pragma unroll
    for (int i = 0; i < 2; ++i)
      #pragma unroll
      for (int j = 0; j < 2; ++j)
        acc[i][j] = __builtin_amdgcn_wmma_f32_16x16x32_f16(
            false, af[i], false, bf[j], (short)0, acc[i][j], false, false);
    __syncthreads();
  }

  #pragma unroll
  for (int i = 0; i < 2; ++i) {
    #pragma unroll
    for (int j = 0; j < 2; ++j) {
      #pragma unroll
      for (int r = 0; r < 8; ++r) {
        int row = wr * 32 + i * 16 + r + ((L & 16) >> 1);
        int m = m0 + row;
        int o = n0 + wc * 32 + j * 16 + (L & 15);
        out[(size_t)m * 256 + o] = acc[i][j][r] + bias[o];
      }
    }
  }
}

// =====================================================================
extern "C" void kernel_launch(void* const* d_in, const int* in_sizes, int n_in,
                              void* d_out, int out_size, void* d_ws, size_t ws_size,
                              hipStream_t stream) {
  const float* q    = (const float*)d_in[0];
  const float* k    = (const float*)d_in[1];
  const float* v    = (const float*)d_in[2];
  const float* Wqkv = (const float*)d_in[3];
  const float* bqkv = (const float*)d_in[4];
  const float* Wout = (const float*)d_in[5];
  const float* bout = (const float*)d_in[6];
  (void)in_sizes; (void)n_in; (void)out_size; (void)ws_size;

  // workspace layout (f16 elements): Qh | Kh | Vh | Oh, each B*H*N*DH = 4M elems
  _Float16* ws = (_Float16*)d_ws;
  const size_t SZ = (size_t)8 * 8 * 2048 * 32;     // 4,194,304
  _Float16* Qh = ws;
  _Float16* Kh = ws + SZ;
  _Float16* Vh = ws + 2 * SZ;
  _Float16* Oh = ws + 3 * SZ;

  // 1) QKV projection: grid = (768/64 col tiles, 16384/128 row tiles)
  qkv_gemm_kernel<<<dim3(12, 128), 256, 0, stream>>>(q, k, v, Wqkv, bqkv, Qh, Kh, Vh);
  // 2) causal flash attention: grid = (B*H, N/64)
  flash_attn_kernel<<<dim3(64, 32), 128, 0, stream>>>(Qh, Kh, Vh, Oh);
  // 3) output projection: grid = (256/64, 16384/128)
  out_proj_kernel<<<dim3(4, 128), 256, 0, stream>>>(Oh, Wout, bout, (float*)d_out);
}
